// TransformerSeqLayer_10179072491535
// MI455X (gfx1250) — matrix operand order repaired and checked by reading, not compiled, not run on hardware
//
#include <hip/hip_runtime.h>
#include <stdint.h>

typedef unsigned short u16;
typedef unsigned int   u32;
typedef unsigned long long u64;
typedef __attribute__((ext_vector_type(16))) __bf16 v16bf;
typedef __attribute__((ext_vector_type(8)))  float  v8f;
typedef __attribute__((ext_vector_type(4)))  unsigned int u32x4;
typedef __attribute__((ext_vector_type(8)))  int i32x8;
typedef __attribute__((ext_vector_type(4)))  int i32x4;

union Frag16 { v16bf bf; u32 w[8]; u16 h[16]; };

// ---- problem constants (from reference) ----
#define BB   4
#define MM   1024
#define HH   1024
#define KH   16
#define DD   64
#define LL   1024
#define EE   8
#define DHD  4096
#define TT   (BB*MM)    // 4096 tokens
#define TKV  (LL+MM)    // 2048 kv positions per batch
#define BKH  (BB*KH)    // 64 (batch*heads)

__device__ __forceinline__ u16 f32_to_bf16(float f){
  u32 u = __float_as_uint(f);
  u32 r = u + 0x7FFFu + ((u >> 16) & 1u);   // round-to-nearest-even
  return (u16)(r >> 16);
}
__device__ __forceinline__ float bf16_to_f32(u16 h){
  return __uint_as_float(((u32)h) << 16);
}

__device__ __forceinline__ v8f wmma_bf16(const Frag16& a, const Frag16& b, v8f c){
  return __builtin_amdgcn_wmma_f32_16x16x32_bf16(false, a.bf, false, b.bf,
                                                 (short)0, c, false, false);
}

// ---- Tensor Data Mover: DMA a 2D tile (row-major, 2-byte elems) into LDS ----
// pad_interval_log: LDS pad of 1 dword after every 2^(pad_interval_log+1) dwords.
__device__ __forceinline__ void tdm_load_tile_2d(u32 lds_byte_off, u64 gaddr,
                                                 u32 tile_elems_x, u32 tile_rows,
                                                 u32 stride_elems, u32 pad_interval_log)
{
  const u32 td0 = 1u << 20, td1 = 1u << 20;   // large tensor dims (no OOB clipping)
  u32x4 g0;
  g0[0] = 1u;                                  // count=1, user descriptor
  g0[1] = lds_byte_off;                        // D#.lds_addr
  g0[2] = (u32)gaddr;                          // D#.global_addr[31:0]
  g0[3] = (u32)(gaddr >> 32) | (2u << 30);     // addr[56:32] | type=2 ("image")
  i32x8 g1;
  g1[0] = (int)((1u << 16) | (1u << 20) | (pad_interval_log << 22)); // 2B elems, pad_en
  g1[1] = (int)((td0 & 0xFFFFu) << 16);                              // tensor_dim0 lo
  g1[2] = (int)(((td0 >> 16) & 0xFFFFu) | ((td1 & 0xFFFFu) << 16));  // td0 hi | td1 lo
  g1[3] = (int)(((td1 >> 16) & 0xFFFFu) | (tile_elems_x << 16));     // td1 hi | tile_dim0
  g1[4] = (int)(tile_rows & 0xFFFFu);                                // tile_dim1 (dim2=0)
  g1[5] = (int)stride_elems;                                         // dim0 stride lo
  g1[6] = 0;                                                         // stride hi | d1s lo
  g1[7] = 0;
  i32x4 gz = {0,0,0,0};
#if defined(__clang_major__) && (__clang_major__ >= 23)
  i32x8 gz8 = {0,0,0,0,0,0,0,0};
  __builtin_amdgcn_tensor_load_to_lds(g0, g1, gz, gz, gz8, 0);
#else
  __builtin_amdgcn_tensor_load_to_lds(g0, g1, gz, gz, 0);
#endif
}

// ===================== utility kernels =====================

__global__ __launch_bounds__(256)
void convert_bf16_kernel(const float* __restrict__ in, u16* __restrict__ out, size_t n){
  size_t i = (size_t)blockIdx.x*blockDim.x + threadIdx.x;
  size_t stride = (size_t)gridDim.x*blockDim.x;
  for (; i < n; i += stride) out[i] = f32_to_bf16(in[i]);
}

__global__ __launch_bounds__(256)
void concat_bf16_kernel(const float* __restrict__ cache, const float* __restrict__ h,
                        u16* __restrict__ out){
  size_t n = (size_t)BB*TKV*HH;
  size_t i = (size_t)blockIdx.x*blockDim.x + threadIdx.x;
  size_t stride = (size_t)gridDim.x*blockDim.x;
  for (; i < n; i += stride){
    size_t b   = i / ((size_t)TKV*HH);
    size_t rem = i % ((size_t)TKV*HH);
    size_t r   = rem / HH;
    size_t c   = rem % HH;
    float val = (r < LL) ? cache[(b*LL + r)*HH + c]
                         : h[(b*MM + (r - LL))*HH + c];
    out[i] = f32_to_bf16(val);
  }
}

__global__ __launch_bounds__(256)
void zero_f32_kernel(float* __restrict__ p, size_t n){
  size_t i = (size_t)blockIdx.x*blockDim.x + threadIdx.x;
  size_t stride = (size_t)gridDim.x*blockDim.x;
  for (; i < n; i += stride) p[i] = 0.f;
}

// ===================== bf16 WMMA GEMM with TDM tile staging =====================
// C[MxN] = A[MxK] @ B[KxN] (row major, bf16 inputs, fp32 accumulate)
// Block tile 128x128x32, 8 waves (2x4), each wave 64x32 = 8 wmma tiles.
// A tile: TDM -> LDS [m][k], padded stride 34 u16 (1 dword pad / 16 dwords).
// B tile: TDM -> LDS [k][n], padded stride 130 u16 (1 dword pad / 64 dwords).

constexpr int EPI_F32           = 0; // Cf[m][n] = acc
constexpr int EPI_BIAS_RELU_B16 = 1; // Cb[m][n] = bf16(relu(acc + bias[n]))
constexpr int EPI_BIAS_SCALE_AC = 2; // Cf[m][n] += rowScale[m*stride+off]*(acc+bias[n])
constexpr int EPI_HEADS_B16     = 3; // split heads: Cb[(b*KH+kh)*rowsPerB+m][d]

template<int EPI>
__global__ __launch_bounds__(256)
void gemm_bf16_kernel(const u16* __restrict__ A, const u16* __restrict__ Bm,
                      const float* __restrict__ bias,
                      float* __restrict__ Cf, u16* __restrict__ Cb,
                      const float* __restrict__ rowScale, int scaleStride, int scaleOff,
                      int M, int N, int Kd, int rowsPerB)
{
  __shared__ u16 As[128*34];   // [m][k]
  __shared__ u16 Bs[32*130];   // [k][n]
  const int tid  = threadIdx.x;
  const int lane = tid & 31;
  const int wave = tid >> 5;
  const int row0 = blockIdx.x * 128;
  const int col0 = blockIdx.y * 128;
  const int wr   = wave >> 2;      // 0..1
  const int wc   = wave & 3;       // 0..3
  const int hl   = lane >> 4;      // lane half
  const int lr   = lane & 15;

  const u32 asOff = (u32)(size_t)(void*)As;   // LDS byte address (generic low 32b)
  const u32 bsOff = (u32)(size_t)(void*)Bs;

  v8f acc[4][2];
  #pragma unroll
  for (int i=0;i<4;i++)
    #pragma unroll
    for (int j=0;j<2;j++)
      #pragma unroll
      for (int v=0;v<8;v++) acc[i][j][v] = 0.f;

  for (int k0 = 0; k0 < Kd; k0 += 32){
    if (wave == 0){
      // A tile: 128 rows x 32 elems, row stride Kd
      tdm_load_tile_2d(asOff, (u64)(uintptr_t)(A + (size_t)row0*Kd + k0),
                       32u, 128u, (u32)Kd, 3u);
      // B tile: 32 rows x 128 elems, row stride N
      tdm_load_tile_2d(bsOff, (u64)(uintptr_t)(Bm + (size_t)k0*N + col0),
                       128u, 32u, (u32)N, 5u);
      __builtin_amdgcn_s_wait_tensorcnt(0);
    }
    __syncthreads();

    Frag16 af[4], bfr[2];
    #pragma unroll
    for (int i=0;i<4;i++){
      int m = wr*64 + i*16 + lr;
      #pragma unroll
      for (int v=0;v<8;v++){
        int kk = hl*16 + v*2;   // A 16x32 layout
        af[i].w[v] = *(const u32*)&As[m*34 + kk];
      }
    }
    #pragma unroll
    for (int j=0;j<2;j++){
      int n = wc*32 + j*16 + lr;
      #pragma unroll
      for (int v=0;v<8;v++){
        int kk = hl*16 + v*2;                    // B 32x16 layout (per-lane column)
        u32 lo = Bs[kk*130 + n];
        u32 hi = Bs[(kk+1)*130 + n];
        bfr[j].w[v] = lo | (hi << 16);
      }
    }
    #pragma unroll
    for (int i=0;i<4;i++)
      #pragma unroll
      for (int j=0;j<2;j++)
        acc[i][j] = wmma_bf16(af[i], bfr[j], acc[i][j]);
    __syncthreads();
  }

  // epilogue (C layout: vgpr v -> row hl*8+v, col lr)
  #pragma unroll
  for (int i=0;i<4;i++)
    #pragma unroll
    for (int j=0;j<2;j++){
      int nb = col0 + wc*32 + j*16 + lr;
      #pragma unroll
      for (int v=0;v<8;v++){
        int mb = row0 + wr*64 + i*16 + hl*8 + v;
        float val = acc[i][j][v];
        if (EPI == EPI_F32){
          Cf[(size_t)mb*N + nb] = val;
        } else if (EPI == EPI_BIAS_RELU_B16){
          val += bias[nb];
          val = val > 0.f ? val : 0.f;
          Cb[(size_t)mb*N + nb] = f32_to_bf16(val);
        } else if (EPI == EPI_BIAS_SCALE_AC){
          val += bias[nb];
          float s = rowScale[(size_t)mb*scaleStride + scaleOff];
          Cf[(size_t)mb*N + nb] += s * val;
        } else { // EPI_HEADS_B16
          int b   = mb / rowsPerB;
          int mm2 = mb % rowsPerB;
          int khd = nb >> 6;
          int d   = nb & 63;
          Cb[(((size_t)(b*KH + khd)*rowsPerB + mm2) << 6) + d] = f32_to_bf16(val);
        }
      }
    }
}

// ===================== windowed attention (flash-style, WMMA) =====================
// One wave (32 threads) per (head bk, 16-query tile). Query m attends to
// h_all keys n = m .. m+L-1 with score q.k[n] + q.pe[n-m], softmax over the window.
// q/k/v in head layout: [BKH][rows][64] bf16. Output: bf16 [B*M, H] (merged heads).

#define PSTR 1026

__global__ __launch_bounds__(32)
void attn_kernel(const u16* __restrict__ qb, const u16* __restrict__ kb,
                 const u16* __restrict__ vb, const u16* __restrict__ peb,
                 u16* __restrict__ outb)
{
  const int bk = blockIdx.x >> 6;       // 64 m-tiles per head
  const int mt = blockIdx.x & 63;
  const int m0 = mt * 16;
  const int lane = threadIdx.x;
  const int hl = lane >> 4;
  const int lr = lane & 15;

  __shared__ u16 P[16*PSTR];            // positional scores P[r][l] (bf16)
  __shared__ u16 probs[16*34];          // C-layout -> A-layout round trip

  const u16* qh = qb + (size_t)bk*MM*DD;
  const u16* kh = kb + (size_t)bk*TKV*DD;
  const u16* vh = vb + (size_t)bk*TKV*DD;

  // q tile as two A fragments (K = D = 64)
  Frag16 aq[2];
  #pragma unroll
  for (int f=0; f<2; f++)
    #pragma unroll
    for (int v=0; v<8; v++){
      int d = f*32 + hl*16 + v*2;
      aq[f].w[v] = *(const u32*)&qh[(size_t)(m0 + lr)*DD + d];
    }

  // P = q_tile @ key_pe : 16 x 1024, via WMMA in 16-col chunks
  for (int c=0; c<64; c++){
    int n0 = c*16;
    Frag16 bp[2];
    #pragma unroll
    for (int f=0; f<2; f++)
      #pragma unroll
      for (int v=0; v<8; v++){
        int d = f*32 + hl*16 + v*2;
        u32 lo = peb[(size_t)d*LL + n0 + lr];
        u32 hi = peb[(size_t)(d+1)*LL + n0 + lr];
        bp[f].w[v] = lo | (hi << 16);
      }
    v8f pc = {0.f,0.f,0.f,0.f,0.f,0.f,0.f,0.f};
    pc = wmma_bf16(aq[0], bp[0], pc);
    pc = wmma_bf16(aq[1], bp[1], pc);
    #pragma unroll
    for (int v=0; v<8; v++){
      int r = hl*8 + v;
      P[r*PSTR + n0 + lr] = f32_to_bf16(pc[v]);
    }
  }
  __syncthreads();

  float mi[8], li[8];
  v8f o[4];
  #pragma unroll
  for (int v=0; v<8; v++){ mi[v] = -1e30f; li[v] = 0.f; }
  #pragma unroll
  for (int t=0; t<4; t++)
    #pragma unroll
    for (int v=0; v<8; v++) o[t][v] = 0.f;

  // 33 chunks of 32 keys cover n in [m0, m0+1038]
  for (int c=0; c<33; c++){
    int nbase = m0 + c*32;

    // prefetch next chunk's K/V rows (global_prefetch)
    {
      int pn = nbase + 32 + lane;
      if (pn > TKV-1) pn = TKV-1;
      __builtin_prefetch(&kh[(size_t)pn*DD], 0, 1);
      __builtin_prefetch(&vh[(size_t)pn*DD], 0, 1);
    }

    float pv[2][8];
    float cmax[8];
    #pragma unroll
    for (int v=0; v<8; v++) cmax[v] = -1e30f;

    #pragma unroll
    for (int t2=0; t2<2; t2++){
      int n0  = nbase + t2*16;
      int nld = n0 + lr; if (nld > TKV-1) nld = TKV-1;   // masked anyway
      Frag16 bkf[2];
      #pragma unroll
      for (int f=0; f<2; f++)
        #pragma unroll
        for (int v=0; v<8; v++){
          int d = f*32 + hl*16 + v*2;
          bkf[f].w[v] = *(const u32*)&kh[(size_t)nld*DD + d];
        }
      v8f s = {0.f,0.f,0.f,0.f,0.f,0.f,0.f,0.f};
      s = wmma_bf16(aq[0], bkf[0], s);
      s = wmma_bf16(aq[1], bkf[1], s);
      #pragma unroll
      for (int v=0; v<8; v++){
        int r = hl*8 + v;
        int lrel = n0 + lr - (m0 + r);          // skew: relative position
        float val = -1e30f;
        if (lrel >= 0 && lrel < LL)
          val = (s[v] + bf16_to_f32(P[r*PSTR + lrel])) * 0.125f;  // /sqrt(64)
        pv[t2][v] = val;
        cmax[v] = fmaxf(cmax[v], val);
      }
    }
    // row max over 16 columns (lanes within each half hold same rows)
    #pragma unroll
    for (int mk=1; mk<16; mk<<=1)
      #pragma unroll
      for (int v=0; v<8; v++) cmax[v] = fmaxf(cmax[v], __shfl_xor(cmax[v], mk, 32));

    float corr[8], psum[8];
    #pragma unroll
    for (int v=0; v<8; v++){
      float nm = fmaxf(mi[v], cmax[v]);
      corr[v]  = __expf(mi[v] - nm);
      mi[v]    = nm;
      psum[v]  = 0.f;
    }
    #pragma unroll
    for (int t2=0; t2<2; t2++)
      #pragma unroll
      for (int v=0; v<8; v++){
        float p = __expf(pv[t2][v] - mi[v]);
        pv[t2][v] = p;
        psum[v] += p;
      }
    #pragma unroll
    for (int mk=1; mk<16; mk<<=1)
      #pragma unroll
      for (int v=0; v<8; v++) psum[v] += __shfl_xor(psum[v], mk, 32);
    #pragma unroll
    for (int v=0; v<8; v++) li[v] = li[v]*corr[v] + psum[v];
    #pragma unroll
    for (int t=0; t<4; t++)
      #pragma unroll
      for (int v=0; v<8; v++) o[t][v] *= corr[v];

    // probs: C layout -> LDS -> A layout (16 x 32 keys)
    __syncthreads();
    #pragma unroll
    for (int t2=0; t2<2; t2++)
      #pragma unroll
      for (int v=0; v<8; v++)
        probs[(hl*8 + v)*34 + t2*16 + lr] = f32_to_bf16(pv[t2][v]);
    __syncthreads();

    Frag16 ap;
    #pragma unroll
    for (int v=0; v<8; v++){
      int kk = hl*16 + v*2;
      ap.w[v] = *(const u32*)&probs[lr*34 + kk];
    }
    // O += probs @ V (4 d-chunks of 16)
    #pragma unroll
    for (int t=0; t<4; t++){
      Frag16 bv;
      #pragma unroll
      for (int v=0; v<8; v++){
        int k1 = nbase + hl*16 + v*2;
        int k2 = k1 + 1;
        if (k1 > TKV-1) k1 = TKV-1;
        if (k2 > TKV-1) k2 = TKV-1;
        u32 lo = vh[(size_t)k1*DD + t*16 + lr];
        u32 hi = vh[(size_t)k2*DD + t*16 + lr];
        bv.w[v] = lo | (hi << 16);
      }
      o[t] = wmma_bf16(ap, bv, o[t]);
    }
  }

  // write merged-head output (bf16), channel block = head index
  int b = bk >> 4, khd = bk & 15;
  #pragma unroll
  for (int t=0; t<4; t++)
    #pragma unroll
    for (int v=0; v<8; v++){
      int r = hl*8 + v;
      int d = t*16 + lr;
      float val = o[t][v] / li[v];
      outb[((size_t)(b*MM + m0 + r)*HH) + khd*DD + d] = f32_to_bf16(val);
    }
}

// ===================== add + layernorm =====================

__global__ __launch_bounds__(256)
void add_ln_kernel(const float* __restrict__ a, const float* __restrict__ b,
                   const float* __restrict__ g, const float* __restrict__ be,
                   float* __restrict__ outf, u16* __restrict__ outb, int Hd)
{
  int row = blockIdx.x, tid = threadIdx.x;
  __shared__ float s1[256], s2[256];
  const float* ar = a + (size_t)row*Hd;
  const float* br = b + (size_t)row*Hd;
  float sum = 0.f, sq = 0.f;
  for (int i = tid; i < Hd; i += 256){
    float x = ar[i] + br[i];
    sum += x; sq += x*x;
  }
  s1[tid] = sum; s2[tid] = sq; __syncthreads();
  for (int off = 128; off > 0; off >>= 1){
    if (tid < off){ s1[tid] += s1[tid+off]; s2[tid] += s2[tid+off]; }
    __syncthreads();
  }
  float mu  = s1[0] / Hd;
  float var = s2[0] / Hd - mu*mu;
  float inv = rsqrtf(var + 1e-5f);
  for (int i = tid; i < Hd; i += 256){
    float x = ar[i] + br[i];
    float y = (x - mu) * inv * g[i] + be[i];
    if (outf) outf[(size_t)row*Hd + i] = y;
    if (outb) outb[(size_t)row*Hd + i] = f32_to_bf16(y);
  }
}

// ===================== MoE gate: logits -> top2 -> softmax -> dense score matrix =====

__global__ __launch_bounds__(256)
void gate_kernel(const float* __restrict__ x, const float* __restrict__ gw,
                 const float* __restrict__ gb, float* __restrict__ scoreE, int Hd)
{
  int t = blockIdx.x, tid = threadIdx.x;
  __shared__ float s[256*EE];
  float acc[EE];
  #pragma unroll
  for (int e=0;e<EE;e++) acc[e] = 0.f;
  const float* xr = x + (size_t)t*Hd;
  for (int i = tid; i < Hd; i += 256){
    float xv = xr[i];
    #pragma unroll
    for (int e=0;e<EE;e++) acc[e] += xv * gw[i*EE + e];
  }
  #pragma unroll
  for (int e=0;e<EE;e++) s[tid*EE + e] = acc[e];
  __syncthreads();
  for (int off = 128; off > 0; off >>= 1){
    if (tid < off){
      #pragma unroll
      for (int e=0;e<EE;e++) s[tid*EE + e] += s[(tid+off)*EE + e];
    }
    __syncthreads();
  }
  if (tid == 0){
    float lg[EE];
    #pragma unroll
    for (int e=0;e<EE;e++) lg[e] = s[e] + gb[e];
    int i0 = 0;
    for (int e=1;e<EE;e++) if (lg[e] > lg[i0]) i0 = e;
    int i1 = -1;
    for (int e=0;e<EE;e++) if (e != i0 && (i1 < 0 || lg[e] > lg[i1])) i1 = e;
    float p0 = 1.f, p1 = __expf(lg[i1] - lg[i0]);
    float inv = 1.f / (p0 + p1);
    #pragma unroll
    for (int e=0;e<EE;e++)
      scoreE[(size_t)t*EE + e] = (e == i0) ? p0*inv : (e == i1) ? p1*inv : 0.f;
  }
}

// ===================== host launcher =====================

extern "C" void kernel_launch(void* const* d_in, const int* in_sizes, int n_in,
                              void* d_out, int out_size, void* d_ws, size_t ws_size,
                              hipStream_t stream)
{
  const float* h_in   = (const float*)d_in[0];
  const float* hcache = (const float*)d_in[1];
  const float* keype  = (const float*)d_in[2];
  const float* wq = (const float*)d_in[3];
  const float* wk = (const float*)d_in[4];
  const float* wv = (const float*)d_in[5];
  const float* wo = (const float*)d_in[6];
  const float* n1g = (const float*)d_in[7];
  const float* n1b = (const float*)d_in[8];
  const float* n2g = (const float*)d_in[9];
  const float* n2b = (const float*)d_in[10];
  const float* mlg = (const float*)d_in[11];
  const float* mlb = (const float*)d_in[12];
  const float* gw  = (const float*)d_in[13];
  const float* gb  = (const float*)d_in[14];
  const float* ew1 = (const float*)d_in[15];
  const float* eb1 = (const float*)d_in[16];
  const float* ew2 = (const float*)d_in[17];
  const float* eb2 = (const float*)d_in[18];
  float* out = (float*)d_out;

  char* ws = (char*)d_ws;
  size_t off = 0;
  auto alloc = [&](size_t bytes)->char*{
    char* p = ws + off; off += (bytes + 255) & ~(size_t)255; return p;
  };

  u16* h_bf    = (u16*)alloc((size_t)TT*HH*2);
  u16* hall_bf = (u16*)alloc((size_t)BB*TKV*HH*2);
  u16* pe_bf   = (u16*)alloc((size_t)DD*LL*2);
  u16* wq_bf   = (u16*)alloc((size_t)HH*HH*2);
  u16* wk_bf   = (u16*)alloc((size_t)HH*HH*2);
  u16* wv_bf   = (u16*)alloc((size_t)HH*HH*2);
  u16* wo_bf   = (u16*)alloc((size_t)HH*HH*2);
  u16* ew1_bf  = (u16*)alloc((size_t)EE*HH*DHD*2);
  u16* ew2_bf  = (u16*)alloc((size_t)EE*DHD*HH*2);
  u16* q_bf    = (u16*)alloc((size_t)BKH*MM*DD*2);
  u16* k_bf    = (u16*)alloc((size_t)BKH*TKV*DD*2);
  u16* v_bf    = (u16*)alloc((size_t)BKH*TKV*DD*2);
  u16* attn_bf = (u16*)alloc((size_t)TT*HH*2);
  float* attn_out_f = (float*)alloc((size_t)TT*HH*4);
  float* h1_f       = (float*)alloc((size_t)TT*HH*4);
  u16*   x_bf       = (u16*)alloc((size_t)TT*HH*2);
  float* scoreE     = (float*)alloc((size_t)TT*EE*4);
  u16*   hid_bf     = (u16*)alloc((size_t)TT*DHD*2);
  float* moe_f      = (float*)alloc((size_t)TT*HH*4);
  float* core_f     = (float*)alloc((size_t)TT*HH*4);
  (void)ws_size; (void)in_sizes; (void)n_in; (void)out_size;

  // 1. precision conversion
  convert_bf16_kernel<<<2048, 256, 0, stream>>>(h_in, h_bf, (size_t)TT*HH);
  concat_bf16_kernel <<<2048, 256, 0, stream>>>(hcache, h_in, hall_bf);
  convert_bf16_kernel<<<64,   256, 0, stream>>>(keype, pe_bf, (size_t)DD*LL);
  convert_bf16_kernel<<<1024, 256, 0, stream>>>(wq, wq_bf, (size_t)HH*HH);
  convert_bf16_kernel<<<1024, 256, 0, stream>>>(wk, wk_bf, (size_t)HH*HH);
  convert_bf16_kernel<<<1024, 256, 0, stream>>>(wv, wv_bf, (size_t)HH*HH);
  convert_bf16_kernel<<<1024, 256, 0, stream>>>(wo, wo_bf, (size_t)HH*HH);
  convert_bf16_kernel<<<4096, 256, 0, stream>>>(ew1, ew1_bf, (size_t)EE*HH*DHD);
  convert_bf16_kernel<<<4096, 256, 0, stream>>>(ew2, ew2_bf, (size_t)EE*DHD*HH);

  // 2. QKV projections (write directly in per-head bf16 layout)
  gemm_bf16_kernel<EPI_HEADS_B16><<<dim3(TT/128, HH/128), 256, 0, stream>>>(
      h_bf, wq_bf, nullptr, nullptr, q_bf, nullptr, 0, 0, TT, HH, HH, MM);
  gemm_bf16_kernel<EPI_HEADS_B16><<<dim3((BB*TKV)/128, HH/128), 256, 0, stream>>>(
      hall_bf, wk_bf, nullptr, nullptr, k_bf, nullptr, 0, 0, BB*TKV, HH, HH, TKV);
  gemm_bf16_kernel<EPI_HEADS_B16><<<dim3((BB*TKV)/128, HH/128), 256, 0, stream>>>(
      hall_bf, wv_bf, nullptr, nullptr, v_bf, nullptr, 0, 0, BB*TKV, HH, HH, TKV);

  // 3. windowed relative attention (flash-style, WMMA)
  attn_kernel<<<BKH*(MM/16), 32, 0, stream>>>(q_bf, k_bf, v_bf, pe_bf, attn_bf);

  // 4. output projection + norm1
  gemm_bf16_kernel<EPI_F32><<<dim3(TT/128, HH/128), 256, 0, stream>>>(
      attn_bf, wo_bf, nullptr, attn_out_f, nullptr, nullptr, 0, 0, TT, HH, HH, 0);
  add_ln_kernel<<<TT, 256, 0, stream>>>(h_in, attn_out_f, n1g, n1b, h1_f, x_bf, HH);

  // 5. MoE gate + dense experts with gated accumulation
  gate_kernel<<<TT, 256, 0, stream>>>(h1_f, gw, gb, scoreE, HH);
  zero_f32_kernel<<<2048, 256, 0, stream>>>(moe_f, (size_t)TT*HH);
  for (int e = 0; e < EE; e++){
    gemm_bf16_kernel<EPI_BIAS_RELU_B16><<<dim3(TT/128, DHD/128), 256, 0, stream>>>(
        x_bf, ew1_bf + (size_t)e*HH*DHD, eb1 + (size_t)e*DHD,
        nullptr, hid_bf, nullptr, 0, 0, TT, DHD, HH, 0);
    gemm_bf16_kernel<EPI_BIAS_SCALE_AC><<<dim3(TT/128, HH/128), 256, 0, stream>>>(
        hid_bf, ew2_bf + (size_t)e*DHD*HH, eb2 + (size_t)e*HH,
        moe_f, nullptr, scoreE, EE, e, TT, HH, DHD, 0);
  }

  // 6. final norms
  add_ln_kernel<<<TT, 256, 0, stream>>>(h1_f, moe_f, mlg, mlb, core_f, nullptr, HH);
  add_ln_kernel<<<TT, 256, 0, stream>>>(h1_f, core_f, n2g, n2b, out, nullptr, HH);
}
